// MemoryOptimizedMLA_30571577212970
// MI455X (gfx1250) — compile-verified, matrix operand order checked by reading, not run on hardware
//
#include <hip/hip_runtime.h>

typedef __bf16 bf16_t;
typedef __attribute__((ext_vector_type(16))) __bf16 bf16x16;
typedef __attribute__((ext_vector_type(8)))  float  f32x8;

// Problem constants
constexpr int Bb  = 2;
constexpr int Ss  = 2048;
constexpr int Dd  = 1024;
constexpr int Hh  = 16;
constexpr int BS  = Bb * Ss;   // 4096 token rows

// ---------------------------------------------------------------------------
// WMMA helpers (CDNA5 wave32 layouts, ISA 7.12.2)
// ---------------------------------------------------------------------------
__device__ __forceinline__ f32x8 wmma_bf16(bf16x16 a, bf16x16 b, f32x8 c) {
  return __builtin_amdgcn_wmma_f32_16x16x32_bf16(false, a, false, b,
                                                 (short)0, c, false, false);
}

union frag_cast { uint4 u[2]; bf16x16 v; };

// Fragment load from an LDS tile where the "stripe" dimension is the row:
//  - A-fragment (16x32 MxK): base tile is M-major, row = lane&15.
//  - B-fragment (32x16 KxN): base tile is N-major (transposed), row = lane&15.
// Per lane: elements [h8..h8+7] and [16+h8..23+h8]  ->  two ds_load_b128.
__device__ __forceinline__ bf16x16 ld_frag(const bf16_t* base, int lane,
                                           int ld) {
  const int m  = lane & 15;
  const int h8 = ((lane >> 4) & 1) * 8;
  const bf16_t* row = base + m * ld + h8;
  frag_cast f;
  f.u[0] = *(const uint4*)(row);
  f.u[1] = *(const uint4*)(row + 16);
  return f.v;
}

// ---------------------------------------------------------------------------
// Pack / convert kernels
// ---------------------------------------------------------------------------
__global__ void pack_bf16_kernel(const float* __restrict__ src,
                                 bf16_t* __restrict__ dst,
                                 int rows, int cols, int ldd, int col0) {
  size_t n = (size_t)rows * cols;
  for (size_t i = (size_t)blockIdx.x * 256 + threadIdx.x; i < n;
       i += (size_t)gridDim.x * 256) {
    size_t r = i / cols;
    size_t c = i % cols;
    dst[r * (size_t)ldd + col0 + c] = (bf16_t)src[i];
  }
}

// Transposing pack: src (K x N, f32 row-major) -> dst (N x K bf16), placed at
// output row offset row0.  dst[(row0+c)*ldd + r] = src[r*N + c]
__global__ void pack_bf16_t_kernel(const float* __restrict__ src,
                                   bf16_t* __restrict__ dst,
                                   int rows, int cols, int ldd, int row0) {
  size_t n = (size_t)rows * cols;
  for (size_t i = (size_t)blockIdx.x * 256 + threadIdx.x; i < n;
       i += (size_t)gridDim.x * 256) {
    size_t r = i / cols;
    size_t c = i % cols;
    dst[(size_t)(row0 + c) * ldd + r] = (bf16_t)src[i];
  }
}

__global__ void copy_f32_kernel(const float* __restrict__ src,
                                float* __restrict__ dst, int n) {
  int i = blockIdx.x * 256 + threadIdx.x;
  if (i < n) dst[i] = src[i];
}

// ---------------------------------------------------------------------------
// Generic bf16 WMMA GEMM: C(MxN) = A(MxK) * W(KxN) + bias, W given TRANSPOSED
// (WT is N x K row-major).  BM=128 BN=64 BK=32, 256 threads = 8 waves
// (4 along M, 2 along N), each wave computes a 32x32 tile (2x2 WMMA frags).
// ---------------------------------------------------------------------------
__global__ __launch_bounds__(256) void gemm_bf16_kernel(
    const bf16_t* __restrict__ A, int lda, const bf16_t* __restrict__ WT,
    int ldk, const float* __restrict__ bias, float* __restrict__ Cf,
    bf16_t* __restrict__ Cb, int ldc, int K) {
  constexpr int BM = 128, BN = 64, BK = 32;
  constexpr int LAS = 40;  // padded LDS stride: 20 banks/row -> conflict-free
  __shared__ __align__(16) bf16_t As[BM * LAS];
  __shared__ __align__(16) bf16_t Wt[BN * LAS];  // N-major (transposed) tile

  const int tid  = threadIdx.x;
  const int lane = tid & 31;
  const int wid  = tid >> 5;
  const int wm   = wid & 3;   // wave row
  const int wn   = wid >> 2;  // wave col
  const int m0   = blockIdx.y * BM;
  const int n0   = blockIdx.x * BN;

  // Per-thread staging coordinates
  const int ar0 = tid >> 2;          // A rows handled: ar0, ar0+... (2 chunks)
  const int acv = (tid & 3) * 8;
  const int wr  = tid >> 2;          // W row 0..63
  const int wcv = (tid & 3) * 8;

  f32x8 acc[2][2] = {};

  for (int k0 = 0; k0 < K; k0 += BK) {
    // --- Load all global chunks into registers first (batched loads) ---
    uint4 ga0 = *(const uint4*)(A + (size_t)(m0 + ar0) * lda + k0 + acv);
    uint4 ga1 = *(const uint4*)(A + (size_t)(m0 + 64 + ar0) * lda + k0 + acv);
    uint4 gw  = *(const uint4*)(WT + (size_t)(n0 + wr) * ldk + k0 + wcv);
    if (k0 + BK < K) {
      __builtin_prefetch(A + (size_t)(m0 + (tid >> 1)) * lda + k0 + BK, 0, 1);
      __builtin_prefetch(WT + (size_t)(n0 + (tid & 63)) * ldk + k0 + BK, 0, 1);
    }
    // --- Store to LDS ---
    *(uint4*)(As + ar0 * LAS + acv)        = ga0;
    *(uint4*)(As + (64 + ar0) * LAS + acv) = ga1;
    *(uint4*)(Wt + wr * LAS + wcv)         = gw;
    __syncthreads();

    bf16x16 a0 = ld_frag(As + (wm * 32) * LAS, lane, LAS);
    bf16x16 a1 = ld_frag(As + (wm * 32 + 16) * LAS, lane, LAS);
    bf16x16 b0 = ld_frag(Wt + (wn * 32) * LAS, lane, LAS);
    bf16x16 b1 = ld_frag(Wt + (wn * 32 + 16) * LAS, lane, LAS);
    acc[0][0] = wmma_bf16(a0, b0, acc[0][0]);
    acc[0][1] = wmma_bf16(a0, b1, acc[0][1]);
    acc[1][0] = wmma_bf16(a1, b0, acc[1][0]);
    acc[1][1] = wmma_bf16(a1, b1, acc[1][1]);
    __syncthreads();
  }

  // Epilogue: C layout row = r + 8*(lane>=16), col = lane&15
  const int colq = lane & 15;
  const int rh   = ((lane >> 4) & 1) * 8;
#pragma unroll
  for (int im = 0; im < 2; ++im)
#pragma unroll
    for (int in = 0; in < 2; ++in)
#pragma unroll
      for (int r = 0; r < 8; ++r) {
        int row = m0 + wm * 32 + im * 16 + rh + r;
        int col = n0 + wn * 32 + in * 16 + colq;
        float v = acc[im][in][r] + bias[col];
        if (Cf)
          Cf[(size_t)row * ldc + col] = v;
        else
          Cb[(size_t)row * ldc + col] = (bf16_t)v;
      }
}

// ---------------------------------------------------------------------------
// RoPE + head assembly.
//   tmp1  : BS x 512  bf16  (c_kv | c_q | k_rot_raw)
//   kvcat : BS x 1792 bf16  (k_base(768) | v(1024))
//   qcat  : BS x 1024 bf16  (q_base(768) | q_rot(256))
// Outputs: Q,K in (B,H,S,64);  V TRANSPOSED in (B,H,64,S) so the P@V
// B-fragments in attention are contiguous per lane.
// ---------------------------------------------------------------------------
__global__ __launch_bounds__(256) void rope_assemble_kernel(
    const bf16_t* __restrict__ tmp1, const bf16_t* __restrict__ kvcat,
    const bf16_t* __restrict__ qcat, bf16_t* __restrict__ Qg,
    bf16_t* __restrict__ Kg, bf16_t* __restrict__ Vt) {
  const int bs = blockIdx.x;
  const int b  = bs >> 11;   // / 2048
  const int s  = bs & 2047;
  const float t = (float)s * (1.0f / 40.0f);  // ROPE_SCALE

  for (int idx = threadIdx.x; idx < Hh * 64; idx += 256) {
    const int h = idx >> 6;
    const int d = idx & 63;
    const size_t dst = ((size_t)(b * Hh + h) * Ss + s) * 64 + d;

    // V transposed: (B,H,64,S)
    Vt[((size_t)(b * Hh + h) * 64 + d) * Ss + s] =
        kvcat[(size_t)bs * 1792 + 768 + h * 64 + d];

    float qv, kv;
    if (d < 48) {
      qv = (float)qcat[(size_t)bs * 1024 + h * 48 + d];
      kv = (float)kvcat[(size_t)bs * 1792 + h * 48 + d];
    } else {
      const int j = d - 48;
      const bf16_t* qr = qcat + (size_t)bs * 1024 + 768 + h * 16;
      const bf16_t* kr = tmp1 + (size_t)bs * 512 + 256 + h * 16;
      float qx = (float)qr[j];
      float kx = (float)kr[j];
      if (j < 8) {
        int jj = j & 3;
        float inv_freq = __powf(10000.0f, -((float)(2 * jj)) * 0.125f);
        float ang = t * inv_freq;
        float c = __cosf(ang), sn = __sinf(ang);
        float qp = (j < 4) ? -(float)qr[j + 4] : (float)qr[j - 4];
        float kp = (j < 4) ? -(float)kr[j + 4] : (float)kr[j - 4];
        qv = qx * c + qp * sn;
        kv = kx * c + kp * sn;
      } else {
        qv = qx;
        kv = kx;
      }
    }
    Qg[dst] = (bf16_t)qv;
    Kg[dst] = (bf16_t)kv;
  }
}

// ---------------------------------------------------------------------------
// Flash attention: 128 threads = 4 waves; block handles 64 query rows for one
// (b,h); each wave owns 16 rows.  Online softmax, no score materialization.
// K tile staged key-major (B = K^T contiguous); V tile staged d-major from the
// transposed V so P@V B-frags are contiguous.  All frags: 2x ds_load_b128.
// ---------------------------------------------------------------------------
__global__ __launch_bounds__(128) void mla_flash_attn_kernel(
    const bf16_t* __restrict__ Qg, const bf16_t* __restrict__ Kg,
    const bf16_t* __restrict__ VTg, bf16_t* __restrict__ Og) {
  constexpr int LKS = 72;  // 36 banks/row -> conflict-free
  __shared__ __align__(16) bf16_t Kt[64 * LKS];  // [key][d]
  __shared__ __align__(16) bf16_t Vt[64 * LKS];  // [d][key]
  __shared__ __align__(16) bf16_t Pt[4 * 16 * 64];

  const int tid  = threadIdx.x;
  const int lane = tid & 31;
  const int wv   = tid >> 5;
  const int qblk = blockIdx.x;
  const int h    = blockIdx.y;
  const int b    = blockIdx.z;
  const size_t bh   = (size_t)(b * Hh + h) * Ss * 64;  // Q/K base
  const size_t bhv  = (size_t)(b * Hh + h) * 64;       // VT row base
  const int q0      = qblk * 64 + wv * 16;

  // Load Q A-fragments for this wave's 16 rows (K = 64 -> 2 frags)
  bf16x16 qa[2];
  {
    const int m  = lane & 15;
    const int h8 = ((lane >> 4) & 1) * 8;
    const bf16_t* qrow = Qg + bh + (size_t)(q0 + m) * 64 + h8;
    frag_cast f0, f1;
    f0.u[0] = *(const uint4*)(qrow);
    f0.u[1] = *(const uint4*)(qrow + 16);
    f1.u[0] = *(const uint4*)(qrow + 32);
    f1.u[1] = *(const uint4*)(qrow + 48);
    qa[0] = f0.v;
    qa[1] = f1.v;
  }

  f32x8 o[4] = {};
  float mi[8], li[8];
#pragma unroll
  for (int r = 0; r < 8; ++r) { mi[r] = -1e30f; li[r] = 0.f; }

  bf16_t* Pw = Pt + wv * 16 * 64;
  const int sr  = tid >> 1;        // staging row 0..63
  const int scv = (tid & 1) * 32;  // staging col chunk base (2 x 4 uint4)

  for (int kt = 0; kt < Ss / 64; ++kt) {
    // --- Stage K (key-major) and V (d-major) tiles: batched loads first ---
    uint4 gk[4], gv[4];
#pragma unroll
    for (int c = 0; c < 4; ++c) {
      gk[c] = *(const uint4*)(Kg + bh + (size_t)(kt * 64 + sr) * 64 + scv +
                              c * 8);
      gv[c] = *(const uint4*)(VTg + (bhv + sr) * Ss + kt * 64 + scv + c * 8);
    }
    if (kt + 1 < Ss / 64) {
      __builtin_prefetch(Kg + bh + (size_t)(kt * 64 + 64 + sr) * 64, 0, 1);
      __builtin_prefetch(VTg + (bhv + sr) * Ss + kt * 64 + 64, 0, 1);
    }
#pragma unroll
    for (int c = 0; c < 4; ++c) {
      *(uint4*)(Kt + sr * LKS + scv + c * 8) = gk[c];
      *(uint4*)(Vt + sr * LKS + scv + c * 8) = gv[c];
    }
    __syncthreads();

    // Scores: 16x64 = 4 n-tiles; B = K^T via key-major tile (contiguous)
    f32x8 sc[4];
#pragma unroll
    for (int t4 = 0; t4 < 4; ++t4) {
      f32x8 z = {};
      z = wmma_bf16(qa[0], ld_frag(Kt + (t4 * 16) * LKS, lane, LKS), z);
      z = wmma_bf16(qa[1], ld_frag(Kt + (t4 * 16) * LKS + 32, lane, LKS), z);
      sc[t4] = z;
    }

    // Scale 1/sqrt(64)
#pragma unroll
    for (int t4 = 0; t4 < 4; ++t4)
#pragma unroll
      for (int r = 0; r < 8; ++r) sc[t4][r] *= 0.125f;

    // Row max across frags, then across the 16-lane half (C-layout rows)
    float rmax[8];
#pragma unroll
    for (int r = 0; r < 8; ++r) {
      float v = fmaxf(fmaxf(sc[0][r], sc[1][r]), fmaxf(sc[2][r], sc[3][r]));
      v = fmaxf(v, __shfl_xor(v, 1, 32));
      v = fmaxf(v, __shfl_xor(v, 2, 32));
      v = fmaxf(v, __shfl_xor(v, 4, 32));
      v = fmaxf(v, __shfl_xor(v, 8, 32));
      rmax[r] = v;
    }

    float alpha[8], rsum[8];
#pragma unroll
    for (int r = 0; r < 8; ++r) {
      float mn = fmaxf(mi[r], rmax[r]);
      alpha[r] = __expf(mi[r] - mn);
      mi[r]    = mn;
      rsum[r]  = 0.f;
    }
#pragma unroll
    for (int t4 = 0; t4 < 4; ++t4)
#pragma unroll
      for (int r = 0; r < 8; ++r) {
        float p = __expf(sc[t4][r] - mi[r]);
        sc[t4][r] = p;
        rsum[r] += p;
      }
#pragma unroll
    for (int r = 0; r < 8; ++r) {
      float v = rsum[r];
      v += __shfl_xor(v, 1, 32);
      v += __shfl_xor(v, 2, 32);
      v += __shfl_xor(v, 4, 32);
      v += __shfl_xor(v, 8, 32);
      li[r] = li[r] * alpha[r] + v;
    }
#pragma unroll
    for (int t4 = 0; t4 < 4; ++t4)
#pragma unroll
      for (int r = 0; r < 8; ++r) o[t4][r] *= alpha[r];

    // Re-layout P: C layout -> LDS (row-major 16x64), then load as A frags
    {
      int col = lane & 15;
      int rh  = ((lane >> 4) & 1) * 8;
#pragma unroll
      for (int t4 = 0; t4 < 4; ++t4)
#pragma unroll
        for (int r = 0; r < 8; ++r)
          Pw[(rh + r) * 64 + t4 * 16 + col] = (bf16_t)sc[t4][r];
    }
    bf16x16 pa0 = ld_frag(Pw, lane, 64);       // keys 0..31
    bf16x16 pa1 = ld_frag(Pw + 32, lane, 64);  // keys 32..63

    // O += P @ V  (V tile is d-major => B-frags contiguous)
#pragma unroll
    for (int t4 = 0; t4 < 4; ++t4) {
      o[t4] = wmma_bf16(pa0, ld_frag(Vt + (t4 * 16) * LKS, lane, LKS), o[t4]);
      o[t4] =
          wmma_bf16(pa1, ld_frag(Vt + (t4 * 16) * LKS + 32, lane, LKS), o[t4]);
    }
    __syncthreads();
  }

  // Normalize and store (attn matrix row-major BS x 1024, bf16)
  const int col = lane & 15;
  const int rh  = ((lane >> 4) & 1) * 8;
#pragma unroll
  for (int t4 = 0; t4 < 4; ++t4)
#pragma unroll
    for (int r = 0; r < 8; ++r) {
      float v = o[t4][r] / li[r];
      size_t row = (size_t)b * Ss + qblk * 64 + wv * 16 + rh + r;
      Og[row * 1024 + h * 64 + t4 * 16 + col] = (bf16_t)v;
    }
}

// ---------------------------------------------------------------------------
// Host-side orchestration
// ---------------------------------------------------------------------------
extern "C" void kernel_launch(void* const* d_in, const int* in_sizes, int n_in,
                              void* d_out, int out_size, void* d_ws,
                              size_t ws_size, hipStream_t stream) {
  const float* h_f    = (const float*)d_in[0];
  const float* W_dkv  = (const float*)d_in[1];
  const float* b_dkv  = (const float*)d_in[2];
  const float* W_dq   = (const float*)d_in[3];
  const float* b_dq   = (const float*)d_in[4];
  const float* W_uk   = (const float*)d_in[5];
  const float* b_uk   = (const float*)d_in[6];
  const float* W_uv   = (const float*)d_in[7];
  const float* b_uv   = (const float*)d_in[8];
  const float* W_uq   = (const float*)d_in[9];
  const float* b_uq   = (const float*)d_in[10];
  const float* W_qr   = (const float*)d_in[11];
  const float* b_qr   = (const float*)d_in[12];
  const float* W_kr   = (const float*)d_in[13];
  const float* b_kr   = (const float*)d_in[14];
  const float* W_o    = (const float*)d_in[15];
  const float* b_o    = (const float*)d_in[16];
  float* out          = (float*)d_out;

  char* ws = (char*)d_ws;
  size_t off = 0;
  auto alloc = [&](size_t bytes) -> char* {
    char* p = ws + off;
    off += (bytes + 255) & ~(size_t)255;
    return p;
  };

  bf16_t* hbf    = (bf16_t*)alloc((size_t)BS * Dd * 2);      // 4096x1024
  bf16_t* wcat1T = (bf16_t*)alloc((size_t)512 * 1024 * 2);   // (dkv|dq|kr)^T
  bf16_t* wcat2T = (bf16_t*)alloc((size_t)1792 * 128 * 2);   // (uk|uv)^T
  bf16_t* wcat3T = (bf16_t*)alloc((size_t)1024 * 128 * 2);   // (uq|qr)^T
  bf16_t* woT    = (bf16_t*)alloc((size_t)1024 * 1024 * 2);  // W_o^T
  float*  bcat1  = (float*)alloc(512 * 4);
  float*  bcat2  = (float*)alloc(1792 * 4);
  float*  bcat3  = (float*)alloc(1024 * 4);
  bf16_t* tmp1   = (bf16_t*)alloc((size_t)BS * 512 * 2);     // ckv|cq|krot
  bf16_t* kvcat  = (bf16_t*)alloc((size_t)BS * 1792 * 2);    // kbase|v
  bf16_t* qcat   = (bf16_t*)alloc((size_t)BS * 1024 * 2);    // qbase|qrot
  bf16_t* Qa     = (bf16_t*)alloc((size_t)Bb * Hh * Ss * 64 * 2);
  bf16_t* Ka     = (bf16_t*)alloc((size_t)Bb * Hh * Ss * 64 * 2);
  bf16_t* VaT    = (bf16_t*)alloc((size_t)Bb * Hh * 64 * Ss * 2);
  bf16_t* attnbf = (bf16_t*)alloc((size_t)BS * 1024 * 2);

  auto packs = [&](const float* s, bf16_t* dst, int rows, int cols, int ldd,
                   int col0) {
    size_t n = (size_t)rows * cols;
    pack_bf16_kernel<<<(int)((n + 255) / 256), 256, 0, stream>>>(s, dst, rows,
                                                                 cols, ldd,
                                                                 col0);
  };
  auto packsT = [&](const float* s, bf16_t* dst, int rows, int cols, int ldd,
                    int row0) {
    size_t n = (size_t)rows * cols;
    pack_bf16_t_kernel<<<(int)((n + 255) / 256), 256, 0, stream>>>(s, dst,
                                                                   rows, cols,
                                                                   ldd, row0);
  };
  auto copyb = [&](const float* s, float* dst, int n) {
    copy_f32_kernel<<<(n + 255) / 256, 256, 0, stream>>>(s, dst, n);
  };

  // Convert / concat (weights transposed to N x K for contiguous B-frags)
  packs (h_f,   hbf,    BS,   1024, 1024, 0);
  packsT(W_dkv, wcat1T, 1024, 128,  1024, 0);
  packsT(W_dq,  wcat1T, 1024, 128,  1024, 128);
  packsT(W_kr,  wcat1T, 1024, 256,  1024, 256);
  packsT(W_uk,  wcat2T, 128,  768,  128,  0);
  packsT(W_uv,  wcat2T, 128,  1024, 128,  768);
  packsT(W_uq,  wcat3T, 128,  768,  128,  0);
  packsT(W_qr,  wcat3T, 128,  256,  128,  768);
  packsT(W_o,   woT,    1024, 1024, 1024, 0);
  copyb(b_dkv, bcat1 + 0,   128);
  copyb(b_dq,  bcat1 + 128, 128);
  copyb(b_kr,  bcat1 + 256, 256);
  copyb(b_uk,  bcat2 + 0,   768);
  copyb(b_uv,  bcat2 + 768, 1024);
  copyb(b_uq,  bcat3 + 0,   768);
  copyb(b_qr,  bcat3 + 768, 256);

  // 1) tmp1 = h @ (W_dkv|W_dq|W_kr)           4096x512, K=1024
  gemm_bf16_kernel<<<dim3(512 / 64, BS / 128), 256, 0, stream>>>(
      hbf, 1024, wcat1T, 1024, bcat1, nullptr, tmp1, 512, 1024);
  // 2) kvcat = c_kv @ (W_uk|W_uv)             4096x1792, K=128
  gemm_bf16_kernel<<<dim3(1792 / 64, BS / 128), 256, 0, stream>>>(
      tmp1, 512, wcat2T, 128, bcat2, nullptr, kvcat, 1792, 128);
  // 3) qcat = c_q @ (W_uq|W_qr)               4096x1024, K=128
  gemm_bf16_kernel<<<dim3(1024 / 64, BS / 128), 256, 0, stream>>>(
      tmp1 + 128, 512, wcat3T, 128, bcat3, nullptr, qcat, 1024, 128);

  // 4) RoPE + assemble Q,K (B,H,S,64) and V^T (B,H,64,S)
  rope_assemble_kernel<<<BS, 256, 0, stream>>>(tmp1, kvcat, qcat, Qa, Ka, VaT);

  // 5) Flash attention -> attnbf (BS x 1024 bf16)
  mla_flash_attn_kernel<<<dim3(Ss / 64, Hh, Bb), 128, 0, stream>>>(Qa, Ka, VaT,
                                                                   attnbf);

  // 6) out = attn @ W_o + b_o                 4096x1024, K=1024, f32 out
  gemm_bf16_kernel<<<dim3(1024 / 64, BS / 128), 256, 0, stream>>>(
      attnbf, 1024, woT, 1024, b_o, out, nullptr, 1024, 1024);
}
